// SwinTransformerBlock_33621003993407
// MI455X (gfx1250) — compile-verified, matrix operand order checked by reading, not run on hardware
//
#include <hip/hip_runtime.h>
#include <hip/hip_bf16.h>

typedef __attribute__((ext_vector_type(16))) _Float16 v16h;
typedef __attribute__((ext_vector_type(8)))  _Float16 v8h;
typedef __attribute__((ext_vector_type(8)))  float    v8f;

namespace {
constexpr int kB     = 64;   // batch
constexpr int kImg   = 56;   // H == W
constexpr int kC     = 96;   // channels
constexpr int kWin   = 7;    // window
constexpr int kShift = 3;
constexpr int kHeads = 3;
constexpr int kHd    = 32;   // head dim
constexpr int kN     = 49;   // tokens / window
constexpr int kNp    = 64;   // padded tokens
constexpr int kNw    = 64;   // windows / image
constexpr int kQkvC  = 288;  // 3*C
constexpr int kFF    = 384;  // 4*C
}

static __device__ __forceinline__ v8f wmma16(v16h a, v16h b, v8f c) {
  // v_wmma_f32_16x16x32_f16 : D = A(16x32 f16) * B(32x16 f16) + C(16x16 f32)
  return __builtin_amdgcn_wmma_f32_16x16x32_f16(false, a, false, b, (short)0, c,
                                                false, false);
}

// Fragment from row-major [rows][stride] storage, fragment covers rows
// r0..r0+15, K columns k0..k0+31.  lane&15 = row index, lane>>4 selects the
// 8-wide K half; elements 0..7 -> K=kb..kb+7, elements 8..15 -> K=kb+16..kb+23.
// Two contiguous 16B loads per lane, concatenated without element moves.
static __device__ __forceinline__ v16h frag_rows(const _Float16* base, int stride,
                                                 int r0, int k0, int lane) {
  const _Float16* p = base + (size_t)(r0 + (lane & 15)) * stride + k0 + ((lane >> 4) << 3);
  v8h lo = *(const v8h*)p;
  v8h hi = *(const v8h*)(p + 16);
  return __builtin_shufflevector(lo, hi, 0, 1, 2, 3, 4, 5, 6, 7,
                                 8, 9, 10, 11, 12, 13, 14, 15);
}

// B fragment from pre-packed weights: tiles stored lane-major, each lane's 16
// halves contiguous -> a single 32B load per lane.
// tile index = ntile * nKtiles + ktile ; tile size = 32 lanes * 16 halves.
static __device__ __forceinline__ v16h frag_packed(const _Float16* wp, int nKtiles,
                                                   int ktile, int ntile, int lane) {
  const _Float16* p = wp + ((size_t)(ntile * nKtiles + ktile) << 9) + (lane << 4);
  return *(const v16h*)p;
}

// ---------------------------------------------------------------------------
// Kernel 0: one-shot f32 -> f16 weight conversion + pack into B-fragment
// layout (stays resident in the 192MB L2 across the whole launch).
// ---------------------------------------------------------------------------
static __device__ __forceinline__ void pack_frag(const float* __restrict__ src,
                                                 _Float16* __restrict__ dst,
                                                 int K, int N, int i) {
  const int e    = i & 15;
  const int lane = (i >> 4) & 31;
  const int tile = i >> 9;
  const int nKt  = K >> 5;
  const int kt   = tile % nKt;
  const int nt   = tile / nKt;
  const int k = kt * 32 + ((lane >> 4) << 3) + (e & 7) + ((e >> 3) << 4);
  const int n = nt * 16 + (lane & 15);
  dst[i] = (_Float16)src[(size_t)k * N + n];
}

__global__ void pack_weights_kernel(const float* __restrict__ qw, const float* __restrict__ pw,
                                    const float* __restrict__ f1, const float* __restrict__ f2,
                                    _Float16* qwp, _Float16* pwp,
                                    _Float16* f1p, _Float16* f2p) {
  int i = blockIdx.x * blockDim.x + threadIdx.x;
  if (i < kC * kQkvC) pack_frag(qw, qwp, kC, kQkvC, i);
  if (i < kC * kC)    pack_frag(pw, pwp, kC, kC, i);
  if (i < kC * kFF)   pack_frag(f1, f1p, kC, kFF, i);
  if (i < kFF * kC)   pack_frag(f2, f2p, kFF, kC, i);
}

// ---------------------------------------------------------------------------
// Kernel 1: per-window fused  LN1 -> shift/partition -> QKV -> attn(+bias,+mask)
//           -> softmax -> @V -> proj -> reverse-shift + residual  => x2
// One workgroup (8 wave32) per window; B*nW = 4096 workgroups.
// ---------------------------------------------------------------------------
__global__ __launch_bounds__(256) void swin_attn_kernel(
    const float* __restrict__ x,
    const float* __restrict__ ln1_g, const float* __restrict__ ln1_b,
    const _Float16* __restrict__ qkv_wp, const float* __restrict__ qkv_b,
    const _Float16* __restrict__ proj_wp, const float* __restrict__ proj_b,
    const float* __restrict__ bias_table,
    float* __restrict__ x2) {
  // 60 KB static LDS, phased reuse:
  //  [0,24576)  : Xs [64][96] f16 (ph1-2)      -> Ss [3][64][64] f16 (ph3+)
  //  [24576,..) : Qs [64][96] f16 (ph2-3)      -> Os [64][96] f16 (ph4+)
  //               Ks [64][96], Vt [96][64] (V transposed)
  __shared__ __align__(16) char smem[61440];
  _Float16* Xs = (_Float16*)smem;
  _Float16* Ss = (_Float16*)smem;
  _Float16* Qs = (_Float16*)(smem + 24576);
  _Float16* Ks = Qs + kNp * kC;
  _Float16* Vt = Ks + kNp * kC;  // [kC][kNp]
  _Float16* Os = Qs;

  const int blk  = blockIdx.x;
  const int b    = blk >> 6;
  const int w    = blk & 63;
  const int wr   = w >> 3, wc = w & 7;
  const int tid  = threadIdx.x, wave = tid >> 5, lane = tid & 31;
  const float scale = 0.17677669529663687f;  // 32^-0.5

  // ---- Phase 1: gather (cyclic -3,-3 roll) + LayerNorm, f16 into LDS ------
  for (int t = wave; t < kNp; t += 8) {
    if (t < kN) {
      const int r = t / kWin, c = t % kWin;
      const int hh = (wr * kWin + r + kShift) % kImg;
      const int ww = (wc * kWin + c + kShift) % kImg;
      const float* px = x + ((size_t)(b * kImg * kImg) + hh * kImg + ww) * kC;
      float v0 = px[lane], v1 = px[lane + 32], v2 = px[lane + 64];
      float s = v0 + v1 + v2;
      float q = v0 * v0 + v1 * v1 + v2 * v2;
#pragma unroll
      for (int m = 16; m >= 1; m >>= 1) {
        s += __shfl_xor(s, m, 32);
        q += __shfl_xor(q, m, 32);
      }
      const float mu   = s * (1.0f / kC);
      const float rstd = rsqrtf(q * (1.0f / kC) - mu * mu + 1e-5f);
      Xs[t * kC + lane]      = (_Float16)((v0 - mu) * rstd * ln1_g[lane]      + ln1_b[lane]);
      Xs[t * kC + lane + 32] = (_Float16)((v1 - mu) * rstd * ln1_g[lane + 32] + ln1_b[lane + 32]);
      Xs[t * kC + lane + 64] = (_Float16)((v2 - mu) * rstd * ln1_g[lane + 64] + ln1_b[lane + 64]);
    } else {  // zero pad rows 49..63
      Xs[t * kC + lane]      = (_Float16)0.f;
      Xs[t * kC + lane + 32] = (_Float16)0.f;
      Xs[t * kC + lane + 64] = (_Float16)0.f;
    }
  }
  __syncthreads();

  // ---- Phase 2: QKV = Xs[64x96] @ W[96x288] (+bias, Q*scale) --------------
  // 72 output tiles (4x18), 9 per wave, 3 WMMA (K=32) each.
  // Tiles never straddle the Q/K/V boundary (96 = 6 n-tiles each).
#pragma unroll 1
  for (int i = 0; i < 9; ++i) {
    const int tt = wave + 8 * i;
    const int tm = tt & 3, tn = tt >> 2;
    v8f acc = {};
#pragma unroll
    for (int ks = 0; ks < 3; ++ks) {
      v16h a  = frag_rows(Xs, kC, tm * 16, ks * 32, lane);
      v16h bf = frag_packed(qkv_wp, 3, ks, tn, lane);
      acc = wmma16(a, bf, acc);
    }
    const int col   = tn * 16 + (lane & 15);
    const int which = tn / 6;  // 0=Q 1=K 2=V (uniform per tile)
    const float bi  = qkv_b[col];
    if (which == 2) {
      // V: store transposed so phase 4's B fragment is two ds_load_b128.
      const int ch = col - 2 * kC;  // 0..95
      v8h pk;
#pragma unroll
      for (int e = 0; e < 8; ++e) pk[e] = (_Float16)(acc[e] + bi);
      *(v8h*)(Vt + ch * kNp + tm * 16 + ((lane >> 4) << 3)) = pk;
    } else {
      _Float16* dst  = (which == 0) ? Qs : Ks;
      const int ch   = col % kC;
      const float mul = (which == 0) ? scale : 1.0f;
#pragma unroll
      for (int e = 0; e < 8; ++e) {
        const int m = tm * 16 + ((lane >> 4) << 3) + e;
        dst[m * kC + ch] = (_Float16)((acc[e] + bi) * mul);
      }
    }
  }
  __syncthreads();

  // ---- Phase 3: scores = Q @ K^T + rel-pos bias + shift mask, f16 ---------
  // 48 tiles (3 heads x 4x4), 6 per wave, K=32 -> single WMMA each.
#pragma unroll 1
  for (int i = 0; i < 6; ++i) {
    const int tt = wave + 8 * i;
    const int h = tt >> 4, rem = tt & 15;
    const int tm = rem >> 2, tn = rem & 3;
    v16h a  = frag_rows(Qs + h * kHd, kC, tm * 16, 0, lane);
    v16h bf = frag_rows(Ks + h * kHd, kC, tn * 16, 0, lane);  // B = K^T (rows = key tokens)
    v8f acc = {};
    acc = wmma16(a, bf, acc);
    const int n = tn * 16 + (lane & 15);
#pragma unroll
    for (int e = 0; e < 8; ++e) {
      const int m = tm * 16 + ((lane >> 4) << 3) + e;
      float v = 0.f;
      if (m < kN && n < kN) {
        const int ri = m / kWin, ci = m % kWin, rj = n / kWin, cj = n % kWin;
        const int idx = (ri - rj + kWin - 1) * (2 * kWin - 1) + (ci - cj + kWin - 1);
        const float bias = bias_table[idx * kHeads + h];
        // shift-mask regions on the (unrolled) mask canvas
        const int im = wr * kWin + ri, jm = wc * kWin + ci;
        const int in_ = wr * kWin + rj, jn = wc * kWin + cj;
        const int regm = ((im < 49) ? 0 : (im < 53 ? 3 : 6)) + ((jm < 49) ? 0 : (jm < 53 ? 1 : 2));
        const int regn = ((in_ < 49) ? 0 : (in_ < 53 ? 3 : 6)) + ((jn < 49) ? 0 : (jn < 53 ? 1 : 2));
        v = acc[e] + bias + ((regm != regn) ? -100.0f : 0.0f);
      }
      Ss[h * (kNp * kNp) + m * kNp + n] = (_Float16)v;
    }
  }
  __syncthreads();

  // ---- Softmax over key dim (49 valid cols), in place, f16 probs ----------
  if (tid < kHeads * kNp) {
    const int h = tid >> 6, m = tid & 63;
    _Float16* row = Ss + h * (kNp * kNp) + m * kNp;
    if (m < kN) {
      float mx = -1e30f;
      for (int n = 0; n < kN; ++n) mx = fmaxf(mx, (float)row[n]);
      float sum = 0.f;
      for (int n = 0; n < kN; ++n) sum += __expf((float)row[n] - mx);
      const float inv = 1.0f / sum;
      for (int n = 0; n < kN; ++n) row[n] = (_Float16)(__expf((float)row[n] - mx) * inv);
      for (int n = kN; n < kNp; ++n) row[n] = (_Float16)0.f;
    } else {
      for (int n = 0; n < kNp; ++n) row[n] = (_Float16)0.f;
    }
  }
  __syncthreads();

  // ---- Phase 4: O = P[64x64] @ V[64x32] per head --------------------------
  // 24 tiles (3 heads x 4x2), 3 per wave, 2 WMMA (K=64) each.
  // B fragment: lane&15 = output dim -> row of Vt; elements = key tokens.
#pragma unroll 1
  for (int i = 0; i < 3; ++i) {
    const int tt = wave + 8 * i;
    const int h = tt >> 3, rem = tt & 7;
    const int tm = rem >> 1, tn = rem & 1;
    v8f acc = {};
#pragma unroll
    for (int ks = 0; ks < 2; ++ks) {
      v16h a  = frag_rows(Ss + h * (kNp * kNp), kNp, tm * 16, ks * 32, lane);
      v16h bf = frag_rows(Vt, kNp, h * kHd + tn * 16, ks * 32, lane);
      acc = wmma16(a, bf, acc);
    }
    const int col = h * kHd + tn * 16 + (lane & 15);
#pragma unroll
    for (int e = 0; e < 8; ++e) {
      const int m = tm * 16 + ((lane >> 4) << 3) + e;
      Os[m * kC + col] = (_Float16)acc[e];
    }
  }
  __syncthreads();

  // ---- Phase 5: proj + reverse shift + residual -> x2 ---------------------
  // 24 tiles (4x6), 3 per wave, 3 WMMA each.
#pragma unroll 1
  for (int i = 0; i < 3; ++i) {
    const int tt = wave + 8 * i;
    const int tm = tt / 6, tn = tt % 6;
    v8f acc = {};
#pragma unroll
    for (int ks = 0; ks < 3; ++ks) {
      v16h a  = frag_rows(Os, kC, tm * 16, ks * 32, lane);
      v16h bf = frag_packed(proj_wp, 3, ks, tn, lane);
      acc = wmma16(a, bf, acc);
    }
    const int col = tn * 16 + (lane & 15);
    const float pb = proj_b[col];
#pragma unroll
    for (int e = 0; e < 8; ++e) {
      const int m = tm * 16 + ((lane >> 4) << 3) + e;
      if (m < kN) {
        const int r = m / kWin, c = m % kWin;
        const int hh = (wr * kWin + r + kShift) % kImg;
        const int ww = (wc * kWin + c + kShift) % kImg;
        const size_t idx = ((size_t)(b * kImg * kImg) + hh * kImg + ww) * kC + col;
        x2[idx] = x[idx] + acc[e] + pb;  // first residual
      }
    }
  }
}

// ---------------------------------------------------------------------------
// Kernel 2: fused LN2 -> fc1 -> exact GELU -> fc2 -> residual  => out
// One workgroup per 64 tokens; 3136 workgroups.
// ---------------------------------------------------------------------------
__global__ __launch_bounds__(256) void swin_mlp_kernel(
    const float* __restrict__ x2,
    const float* __restrict__ ln2_g, const float* __restrict__ ln2_b,
    const _Float16* __restrict__ fc1_wp, const float* __restrict__ fc1_b,
    const _Float16* __restrict__ fc2_wp, const float* __restrict__ fc2_b,
    float* __restrict__ out) {
  __shared__ __align__(16) char smem[61440];
  _Float16* Hs = (_Float16*)smem;            // [64][96]  LN'd activations
  _Float16* Gs = (_Float16*)(smem + 12288);  // [64][384] GELU(fc1) activations

  const int tid = threadIdx.x, wave = tid >> 5, lane = tid & 31;
  const size_t tok0 = (size_t)blockIdx.x * 64;

  // ---- LN2 -> f16 ---------------------------------------------------------
  for (int t = wave; t < 64; t += 8) {
    const float* px = x2 + (tok0 + t) * kC;
    float v0 = px[lane], v1 = px[lane + 32], v2 = px[lane + 64];
    float s = v0 + v1 + v2;
    float q = v0 * v0 + v1 * v1 + v2 * v2;
#pragma unroll
    for (int m = 16; m >= 1; m >>= 1) {
      s += __shfl_xor(s, m, 32);
      q += __shfl_xor(q, m, 32);
    }
    const float mu   = s * (1.0f / kC);
    const float rstd = rsqrtf(q * (1.0f / kC) - mu * mu + 1e-5f);
    Hs[t * kC + lane]      = (_Float16)((v0 - mu) * rstd * ln2_g[lane]      + ln2_b[lane]);
    Hs[t * kC + lane + 32] = (_Float16)((v1 - mu) * rstd * ln2_g[lane + 32] + ln2_b[lane + 32]);
    Hs[t * kC + lane + 64] = (_Float16)((v2 - mu) * rstd * ln2_g[lane + 64] + ln2_b[lane + 64]);
  }
  __syncthreads();

  // ---- fc1 + exact GELU: [64x96] @ [96x384] -------------------------------
  // 96 tiles (4x24), 12 per wave, 3 WMMA each.
#pragma unroll 1
  for (int i = 0; i < 12; ++i) {
    const int tt = wave + 8 * i;
    const int tm = tt & 3, tn = tt >> 2;
    v8f acc = {};
#pragma unroll
    for (int ks = 0; ks < 3; ++ks) {
      v16h a  = frag_rows(Hs, kC, tm * 16, ks * 32, lane);
      v16h bf = frag_packed(fc1_wp, 3, ks, tn, lane);
      acc = wmma16(a, bf, acc);
    }
    const int col = tn * 16 + (lane & 15);
    const float bb = fc1_b[col];
#pragma unroll
    for (int e = 0; e < 8; ++e) {
      const int m = tm * 16 + ((lane >> 4) << 3) + e;
      float v = acc[e] + bb;
      v = 0.5f * v * (1.0f + erff(v * 0.70710678118654752f));  // exact GELU
      Gs[m * kFF + col] = (_Float16)v;
    }
  }
  __syncthreads();

  // ---- fc2 + residual: [64x384] @ [384x96] --------------------------------
  // 24 tiles (4x6), 3 per wave, 12 WMMA each.
#pragma unroll 1
  for (int i = 0; i < 3; ++i) {
    const int tt = wave + 8 * i;
    const int tm = tt / 6, tn = tt % 6;
    v8f acc = {};
#pragma unroll
    for (int ks = 0; ks < 12; ++ks) {
      v16h a  = frag_rows(Gs, kFF, tm * 16, ks * 32, lane);
      v16h bf = frag_packed(fc2_wp, 12, ks, tn, lane);
      acc = wmma16(a, bf, acc);
    }
    const int col = tn * 16 + (lane & 15);
    const float bb = fc2_b[col];
#pragma unroll
    for (int e = 0; e < 8; ++e) {
      const int m = tm * 16 + ((lane >> 4) << 3) + e;
      const size_t row = tok0 + m;
      out[row * kC + col] = x2[row * kC + col] + acc[e] + bb;  // second residual
    }
  }
}

// ---------------------------------------------------------------------------
extern "C" void kernel_launch(void* const* d_in, const int* in_sizes, int n_in,
                              void* d_out, int out_size, void* d_ws, size_t ws_size,
                              hipStream_t stream) {
  const float* x          = (const float*)d_in[0];
  const float* ln1_g      = (const float*)d_in[1];
  const float* ln1_b      = (const float*)d_in[2];
  const float* qkv_w      = (const float*)d_in[3];
  const float* qkv_b      = (const float*)d_in[4];
  const float* proj_w     = (const float*)d_in[5];
  const float* proj_b     = (const float*)d_in[6];
  const float* bias_table = (const float*)d_in[7];
  const float* ln2_g      = (const float*)d_in[8];
  const float* ln2_b      = (const float*)d_in[9];
  const float* fc1_w      = (const float*)d_in[10];
  const float* fc1_b      = (const float*)d_in[11];
  const float* fc2_w      = (const float*)d_in[12];
  const float* fc2_b      = (const float*)d_in[13];

  char* ws = (char*)d_ws;
  float* x2 = (float*)ws;  // first-residual activations [B, H*W, C] f32
  size_t o = (size_t)kB * kImg * kImg * kC * sizeof(float);
  _Float16* qkvp  = (_Float16*)(ws + o); o += (size_t)kC * kQkvC * sizeof(_Float16);
  _Float16* projp = (_Float16*)(ws + o); o += (size_t)kC * kC    * sizeof(_Float16);
  _Float16* f1p   = (_Float16*)(ws + o); o += (size_t)kC * kFF   * sizeof(_Float16);
  _Float16* f2p   = (_Float16*)(ws + o); o += (size_t)kFF * kC   * sizeof(_Float16);

  pack_weights_kernel<<<(kC * kFF + 255) / 256, 256, 0, stream>>>(
      qkv_w, proj_w, fc1_w, fc2_w, qkvp, projp, f1p, f2p);

  swin_attn_kernel<<<kB * kNw, 256, 0, stream>>>(
      x, ln1_g, ln1_b, qkvp, qkv_b, projp, proj_b, bias_table, x2);

  swin_mlp_kernel<<<(kB * kImg * kImg) / 64, 256, 0, stream>>>(
      x2, ln2_g, ln2_b, f1p, fc1_b, f2p, fc2_b, (float*)d_out);
}
